// SinkhornDistance_62620623176409
// MI455X (gfx1250) — matrix operand order repaired and checked by reading, not compiled, not run on hardware
//
#include <hip/hip_runtime.h>

#define NB 16
#define P  1024
#define D  64
#define EPS 0.1f
#define INV_EPS 10.0f
#define MAX_ITER 100
#define THRESH 0.1f
// (1/eps) * log2(e): converts (a-b)/eps into base-2 exponent units
#define SCALE2 14.426950408889634f
#define LN2 0.6931471805599453f
// ln(1/1024 + 1e-8)
#define LOG_MU -6.9314615656f

// native v_exp_f32 is base-2
#define EXP2F(x) __builtin_amdgcn_exp2f(x)

typedef __attribute__((ext_vector_type(16))) _Float16 v16h;
typedef __attribute__((ext_vector_type(8)))  _Float16 v8h;
typedef __attribute__((ext_vector_type(8)))  float    v8f;
typedef __attribute__((ext_vector_type(4)))  float    v4f;

// ---------- normalization stage 1: per (n,d) constant  c = mean / sqrt(var+1e-4)
__global__ void norm_stats(const float* __restrict__ t, float* __restrict__ c) {
  int n = blockIdx.x >> 6, d = blockIdx.x & 63;
  const float* base = t + (size_t)n * P * D + d;
  float s = 0.f, ss = 0.f;
  for (int p = threadIdx.x; p < P; p += 256) {
    float x = base[(size_t)p * D];
    s += x; ss += x * x;
  }
  __shared__ float sm[256], sm2[256];
  sm[threadIdx.x] = s; sm2[threadIdx.x] = ss;
  __syncthreads();
  for (int off = 128; off; off >>= 1) {
    if (threadIdx.x < off) {
      sm[threadIdx.x]  += sm[threadIdx.x + off];
      sm2[threadIdx.x] += sm2[threadIdx.x + off];
    }
    __syncthreads();
  }
  if (threadIdx.x == 0) {
    float sum = sm[0], sumsq = sm2[0];
    float m   = sum / (float)P;
    float var = (sumsq - sum * m) / (float)(P - 1);   // unbiased (ddof=1)
    c[n * D + d] = m * rsqrtf(var + 1e-4f);
  }
}

// ---------- normalization stage 2: xn = t - c ; emit f16 copy + squared row norms
__global__ void norm_apply(const float* __restrict__ t, const float* __restrict__ c,
                           _Float16* __restrict__ th, float* __restrict__ sq) {
  int wave = (blockIdx.x * blockDim.x + threadIdx.x) >> 5;
  int lane = threadIdx.x & 31;
  int n = wave >> 10, p = wave & 1023;
  const float* row  = t  + ((size_t)n * P + p) * D;
  _Float16*    rowh = th + ((size_t)n * P + p) * D;
  float acc = 0.f;
  for (int d = lane; d < D; d += 32) {
    float xn = row[d] - c[n * D + d];
    rowh[d] = (_Float16)xn;
    acc += xn * xn;
  }
  for (int off = 16; off; off >>= 1) acc += __shfl_xor(acc, off, 32);
  if (lane == 0) sq[n * P + p] = acc;
}

// ---------- C = x2 + y2 - 2 * X*Y^T via v_wmma_f32_16x16x32_f16 (one 16x16 tile per wave)
__global__ void cost_gemm(const _Float16* __restrict__ xh, const _Float16* __restrict__ yh,
                          const float* __restrict__ x2, const float* __restrict__ y2,
                          float* __restrict__ C) {
  int wave = (blockIdx.x * blockDim.x + threadIdx.x) >> 5;
  int lane = threadIdx.x & 31;
  int b    = wave >> 12;
  int tile = wave & 4095;
  int tm = tile >> 6, tn = tile & 63;
  int hl  = lane >> 4;     // half-wave (0/1)
  int l16 = lane & 15;
  const _Float16* pA = xh + ((size_t)b * P + tm * 16 + l16) * D;  // A row = M
  const _Float16* pB = yh + ((size_t)b * P + tn * 16 + l16) * D;  // B "row" = N
  v8f acc = {};
#pragma unroll
  for (int k0 = 0; k0 < D; k0 += 32) {
    // A k order used here (contiguous; B matches): lanes0-15 K = k0+0..15, lanes16-31 K = k0+16..31
    v8h alo = *(const v8h*)(pA + k0 + hl * 16);
    v8h ahi = *(const v8h*)(pA + k0 + hl * 16 + 8);
    v16h a;
#pragma unroll
    for (int e = 0; e < 8; ++e) { a[e] = alo[e]; a[8 + e] = ahi[e]; }
    // B k order used here (same as A): lanes0-15 K = k0+0..15, lanes16-31 K = k0+16..31, N = lane%16
    v16h bm = *(const v16h*)(pB + k0 + hl * 16);
    acc = __builtin_amdgcn_wmma_f32_16x16x32_f16(false, a, false, bm,
                                                 (short)0, acc, false, false);
  }
  float yq = y2[b * P + tn * 16 + l16];
  float* outRow = C + ((size_t)b * P + tm * 16 + hl * 8) * P + tn * 16 + l16;
#pragma unroll
  for (int r = 0; r < 8; ++r) {   // D vgpr r: M = r + hl*8, N = l16
    float xp = x2[b * P + tm * 16 + hl * 8 + r];
    outRow[(size_t)r * P] = xp + yq - 2.0f * acc[r];
  }
}

// ---------- Sinkhorn u-update: wave/row; u_new = eps*(log_mu - LSE_j((v_j - C_ij)/eps))
// LSE done in base-2 (v_exp_f32 is natively base-2); b128 streaming loads.
__global__ void sink_u(const float* __restrict__ C, const float* __restrict__ v,
                       float* __restrict__ u, float* __restrict__ errAcc,
                       const int* __restrict__ done) {
  if (*done) return;
  int wave0 = blockIdx.x * 8;
  int b = wave0 >> 10;
  __shared__ __align__(16) float vs[P];
  for (int j = threadIdx.x; j < P; j += 256) vs[j] = v[b * P + j];
  __syncthreads();
  int wave = wave0 + (threadIdx.x >> 5);
  int lane = threadIdx.x & 31;
  int i = wave & 1023;
  const v4f* row4 = (const v4f*)(C + ((size_t)b * P + i) * P);
  const v4f* vs4  = (const v4f*)vs;
  float m = -__builtin_inff(), s = 0.f;
  for (int jb = lane; jb < P / 4; jb += 32) {
    v4f cv = row4[jb];
    v4f vv = vs4[jb];
#pragma unroll
    for (int k = 0; k < 4; ++k) {
      float val = (vv[k] - cv[k]) * SCALE2;
      float nm = fmaxf(m, val);
      s = s * EXP2F(m - nm) + EXP2F(val - nm);
      m = nm;
    }
  }
  for (int off = 16; off; off >>= 1) {
    float m2 = __shfl_xor(m, off, 32);
    float s2 = __shfl_xor(s, off, 32);
    float nm = fmaxf(m, m2);
    s = s * EXP2F(m - nm) + s2 * EXP2F(m2 - nm);
    m = nm;
  }
  if (lane == 0) {
    float lse = LN2 * (m + __log2f(s));       // back to natural log
    float un  = EPS * (LOG_MU - lse);
    float old = u[b * P + i];
    u[b * P + i] = un;
    atomicAdd(errAcc, fabsf(un - old));
  }
}

// ---------- Sinkhorn v-update: thread/column; v_new = eps*(log_nu - LSE_i((u_i - C_ij)/eps))
__global__ void sink_v(const float* __restrict__ C, const float* __restrict__ u,
                       float* __restrict__ v, const int* __restrict__ done) {
  if (*done) return;
  int b   = blockIdx.x >> 2;
  int col = ((blockIdx.x & 3) << 8) + threadIdx.x;
  __shared__ float us[P];
  for (int i = threadIdx.x; i < P; i += 256) us[i] = u[b * P + i];
  __syncthreads();
  const float* base = C + (size_t)b * P * P + col;
  float m = -__builtin_inff(), s = 0.f;
  for (int i = 0; i < P; i += 4) {
#pragma unroll
    for (int k = 0; k < 4; ++k) {
      float val = (us[i + k] - base[(size_t)(i + k) * P]) * SCALE2;
      float nm = fmaxf(m, val);
      s = s * EXP2F(m - nm) + EXP2F(val - nm);
      m = nm;
    }
  }
  v[b * P + col] = EPS * (LOG_MU - LN2 * (m + __log2f(s)));
}

// ---------- convergence check (applied AFTER the iteration, matching reference freeze)
__global__ void sink_check(float* errAcc, int* done) {
  if (threadIdx.x == 0 && blockIdx.x == 0) {
    if (*errAcc / (float)NB < THRESH) *done = 1;
    *errAcc = 0.f;
  }
}

// ---------- per-call state init
__global__ void init_state(float* u, float* v, float* errAcc, int* done, float* cost) {
  int i = blockIdx.x * blockDim.x + threadIdx.x;
  if (i < NB * P) { u[i] = 0.f; v[i] = 0.f; }
  if (i < NB) cost[i] = 0.f;
  if (i == 0) { *errAcc = 0.f; *done = 0; }
}

// ---------- pi = exp(M), cost_b = sum(pi * C); one float4 per thread per row
__global__ void pi_cost(const float* __restrict__ C, const float* __restrict__ u,
                        const float* __restrict__ v, float* __restrict__ pi,
                        float* __restrict__ cost) {
  int b = blockIdx.x >> 10;
  int i = blockIdx.x & 1023;
  float ui = u[b * P + i];
  const v4f* crow = (const v4f*)(C  + ((size_t)b * P + i) * P);
  v4f*       prow = (v4f*)      (pi + ((size_t)b * P + i) * P);
  const v4f* v4   = (const v4f*)(v + b * P);
  int t = threadIdx.x;                 // 256 threads * 4 = 1024 cols
  v4f cv = crow[t];
  v4f vv = v4[t];
  v4f pv;
  float local = 0.f;
#pragma unroll
  for (int k = 0; k < 4; ++k) {
    float p = EXP2F((ui + vv[k] - cv[k]) * SCALE2);
    pv[k] = p;
    local += p * cv[k];
  }
  prow[t] = pv;
  __shared__ float red[256];
  red[t] = local;
  __syncthreads();
  for (int off = 128; off; off >>= 1) {
    if (t < off) red[t] += red[t + off];
    __syncthreads();
  }
  if (t == 0) atomicAdd(&cost[b], red[0]);
}

extern "C" void kernel_launch(void* const* d_in, const int* in_sizes, int n_in,
                              void* d_out, int out_size, void* d_ws, size_t ws_size,
                              hipStream_t stream) {
  const float* x = (const float*)d_in[0];
  const float* y = (const float*)d_in[1];

  float* out  = (float*)d_out;
  float* cost = out;                       // 16
  float* pi   = out + NB;                  // 16M
  float* C    = pi + (size_t)NB * P * P;   // 16M

  char* ws = (char*)d_ws;
  _Float16* xh = (_Float16*)ws;                    // 2 MB
  _Float16* yh = xh + (size_t)NB * P * D;          // 2 MB
  float* x2 = (float*)(yh + (size_t)NB * P * D);   // 64 KB
  float* y2 = x2 + NB * P;                         // 64 KB
  float* cx = y2 + NB * P;                         // 4 KB
  float* cy = cx + NB * D;                         // 4 KB
  float* u  = cy + NB * D;                         // 64 KB
  float* v  = u + NB * P;                          // 64 KB
  float* errAcc = v + NB * P;
  int*   done   = (int*)(errAcc + 1);

  init_state<<<64, 256, 0, stream>>>(u, v, errAcc, done, cost);
  norm_stats<<<NB * D, 256, 0, stream>>>(x, cx);
  norm_stats<<<NB * D, 256, 0, stream>>>(y, cy);
  norm_apply<<<NB * P / 8, 256, 0, stream>>>(x, cx, xh, x2);
  norm_apply<<<NB * P / 8, 256, 0, stream>>>(y, cy, yh, y2);
  cost_gemm<<<NB * 4096 / 8, 256, 0, stream>>>(xh, yh, x2, y2, C);
  for (int it = 0; it < MAX_ITER; ++it) {
    sink_u<<<NB * P / 8, 256, 0, stream>>>(C, v, u, errAcc, done);
    sink_v<<<NB * 4, 256, 0, stream>>>(C, u, v, done);
    sink_check<<<1, 64, 0, stream>>>(errAcc, done);
  }
  pi_cost<<<NB * P, 256, 0, stream>>>(C, u, v, pi, cost);
}
